// Net2_21543555957112
// MI455X (gfx1250) — matrix operand order repaired and checked by reading, not compiled, not run on hardware
//
#include <hip/hip_runtime.h>

typedef __attribute__((ext_vector_type(16))) _Float16 v16h;
typedef __attribute__((ext_vector_type(8)))  _Float16 v8h;
typedef __attribute__((ext_vector_type(8)))  float    v8f;

#define HDIM 128

// ---------------- degree / norm ----------------
__global__ void k_fill_deg(float* __restrict__ deg, int N) {
  int v = blockIdx.x * blockDim.x + threadIdx.x;
  if (v < N) deg[v] = 1.0f;  // self-loop
}

__global__ void k_accum_deg(const long long* __restrict__ dst, float* __restrict__ deg, int E) {
  int e = blockIdx.x * blockDim.x + threadIdx.x;
  if (e < E) atomicAdd(&deg[dst[e]], 1.0f);
}

// dis[] currently holds deg; convert in-place to deg^{-1/2}; init layer-1 self-loop term.
__global__ void k_make_dis(const float* __restrict__ x, float* __restrict__ dis,
                           float* __restrict__ xa, int N) {
  int v = blockIdx.x * blockDim.x + threadIdx.x;
  if (v < N) {
    float r = rsqrtf(dis[v]);   // deg >= 1 always (self-loop)
    dis[v] = r;
    xa[v]  = r * r * x[v];
  }
}

// scalar edge aggregation: acc[d] += dis[s]*dis[d]*val[s]  (used for layers 1 and 3)
__global__ void k_agg_scalar(const long long* __restrict__ src, const long long* __restrict__ dst,
                             const float* __restrict__ dis, const float* __restrict__ val,
                             float* __restrict__ acc, int E) {
  int e = blockIdx.x * blockDim.x + threadIdx.x;
  if (e < E) {
    long long s = src[e], d = dst[e];
    atomicAdd(&acc[d], dis[s] * dis[d] * val[s]);
  }
}

// ---------------- layer 1 expand: h1 = relu(xa * W1 + b1), stored f16 ----------------
__global__ void k_expand_h1(const float* __restrict__ xa, const float* __restrict__ W1,
                            const float* __restrict__ b1, _Float16* __restrict__ h1h) {
  int v = blockIdx.x;
  int f = threadIdx.x;
  float a = xa[v] * W1[f] + b1[f];
  h1h[(size_t)v * HDIM + f] = (_Float16)(a > 0.0f ? a : 0.0f);
}

// ---------------- pre-swizzle W2 into WMMA B-fragment layout ----------------
// Bf layout: [ct(8)][ks(4)][lane(32)][i(16)] halfs.
// lane<16:  b[i] = W2[(ks*32 +  0 + i)][ct*16 + lane]   (K rows 0..15 of the K32 step)
// lane>=16: b[i] = W2[(ks*32 + 16 + i)][ct*16 + lane-16]
__global__ void k_prep_w2(const float* __restrict__ W2, _Float16* __restrict__ Bf) {
  int t = blockIdx.x * blockDim.x + threadIdx.x;   // 0..1023
  if (t >= 8 * 4 * 32) return;
  int lane = t & 31;
  int ks   = (t >> 5) & 3;
  int n    = (t >> 7) * 16 + (lane & 15);
  int koff = (lane < 16) ? 0 : 16;
  _Float16* dstp = Bf + (size_t)t * 16;
#pragma unroll
  for (int i = 0; i < 16; ++i)
    dstp[i] = (_Float16)W2[(ks * 32 + koff + i) * HDIM + n];
}

// ---------------- WMMA GEMM: t2[N x 128] = h1h[N x 128] (f16) @ W2 (f16), f32 acc ----
__global__ void k_gemm(const _Float16* __restrict__ A, const _Float16* __restrict__ Bf,
                       float* __restrict__ C, int N) {
  const int lane = threadIdx.x & 31;
  const int ct   = threadIdx.x >> 5;           // column tile 0..7 (one wave each)
  const int row0 = blockIdx.x * 16;
  if (row0 + 16 > N) return;                   // uniform per block

  const int m    = lane & 15;
  const int aoff = (lane < 16) ? 0 : 16;       // contiguous K: k = 16h + s
  const _Float16* arow = A + (size_t)(row0 + m) * HDIM;

  union U16 { v16h v; v8h h[2]; };
  v8f c = {};
#pragma unroll
  for (int ks = 0; ks < 4; ++ks) {
    const int k = ks * 32;
    U16 a, b;
    a.h[0] = *(const v8h*)(arow + k + aoff);        // K = k+aoff .. +7
    a.h[1] = *(const v8h*)(arow + k + aoff + 8);    // K = k+aoff+8 .. +15
    const _Float16* bp = Bf + (size_t)(((ct * 4 + ks) * 32 + lane) * 16);
    b.h[0] = *(const v8h*)(bp);
    b.h[1] = *(const v8h*)(bp + 8);
    c = __builtin_amdgcn_wmma_f32_16x16x32_f16(false, a.v, false, b.v,
                                               (short)0, c, false, false);
  }
  // C/D layout: VGPR r -> row (r + (lane<16?0:8)), col = lane&15
  const int hi = (lane < 16) ? 0 : 8;
  float* crow = C + (size_t)row0 * HDIM + ct * 16 + (lane & 15);
#pragma unroll
  for (int r = 0; r < 8; ++r)
    crow[(size_t)(r + hi) * HDIM] = c[r];
}

// ---------------- layer 2 aggregation ----------------
__global__ void k_init_acc2(const float* __restrict__ dis, const float* __restrict__ t2,
                            float* __restrict__ acc, size_t NH) {
  size_t i = (size_t)blockIdx.x * blockDim.x + threadIdx.x;
  if (i < NH) {
    size_t v = i >> 7;
    float r = dis[v];
    acc[i] = r * r * t2[i];   // self-loop term
  }
}

// one wave32 per edge; each lane handles 4 contiguous features (float4)
__global__ void k_agg_vec(const long long* __restrict__ src, const long long* __restrict__ dst,
                          const float* __restrict__ dis, const float* __restrict__ t2,
                          float* __restrict__ acc, int E) {
  int e    = (int)(((size_t)blockIdx.x * blockDim.x + threadIdx.x) >> 5);
  int lane = threadIdx.x & 31;
  if (e >= E) return;
  long long s = src[e], d = dst[e];
  float nrm = dis[s] * dis[d];
  const float4 v = *(const float4*)(t2 + (size_t)s * HDIM + lane * 4);
  float* o = acc + (size_t)d * HDIM + lane * 4;
  atomicAdd(o + 0, v.x * nrm);
  atomicAdd(o + 1, v.y * nrm);
  atomicAdd(o + 2, v.z * nrm);
  atomicAdd(o + 3, v.w * nrm);
}

__global__ void k_relu_bias(float* __restrict__ acc, const float* __restrict__ b, size_t NH) {
  size_t i = (size_t)blockIdx.x * blockDim.x + threadIdx.x;
  if (i < NH) {
    float a = acc[i] + b[i & (HDIM - 1)];
    acc[i] = a > 0.0f ? a : 0.0f;
  }
}

// ---------------- layer 3: s3 = h2 . W3 (wave per node) ----------------
__global__ void k_dot_w3(const float* __restrict__ h2, const float* __restrict__ W3,
                         float* __restrict__ s3, int N) {
  int v    = (int)(((size_t)blockIdx.x * blockDim.x + threadIdx.x) >> 5);
  int lane = threadIdx.x & 31;
  if (v >= N) return;
  const float4 hv = *(const float4*)(h2 + (size_t)v * HDIM + lane * 4);
  const float4 wv = *(const float4*)(W3 + lane * 4);
  float p = hv.x * wv.x + hv.y * wv.y + hv.z * wv.z + hv.w * wv.w;
#pragma unroll
  for (int o = 16; o > 0; o >>= 1) p += __shfl_down(p, o, 32);
  if (lane == 0) s3[v] = p;
}

__global__ void k_init_out(const float* __restrict__ dis, const float* __restrict__ s3,
                           const float* __restrict__ b3, float* __restrict__ out, int N) {
  int v = blockIdx.x * blockDim.x + threadIdx.x;
  if (v < N) {
    float r = dis[v];
    out[v] = r * r * s3[v] + b3[0];
  }
}

// ---------------- host launch ----------------
extern "C" void kernel_launch(void* const* d_in, const int* in_sizes, int n_in,
                              void* d_out, int out_size, void* d_ws, size_t ws_size,
                              hipStream_t stream) {
  (void)n_in; (void)out_size; (void)ws_size;
  const float*     x   = (const float*)d_in[0];
  const long long* ei  = (const long long*)d_in[1];
  const float*     W1  = (const float*)d_in[2];
  const float*     b1  = (const float*)d_in[3];
  const float*     W2  = (const float*)d_in[4];
  const float*     b2  = (const float*)d_in[5];
  const float*     W3  = (const float*)d_in[6];
  const float*     b3  = (const float*)d_in[7];
  float*           out = (float*)d_out;

  const int N = in_sizes[0];
  const int E = in_sizes[1] / 2;
  const long long* src = ei;
  const long long* dst = ei + E;

  char* w = (char*)d_ws;
  auto take = [&](size_t bytes) -> char* {
    char* p = w; w += (bytes + 255) & ~(size_t)255; return p;
  };
  float*    dis  = (float*)take((size_t)N * 4);            // deg -> deg^{-1/2}
  float*    xa   = (float*)take((size_t)N * 4);            // layer-1 scalar agg
  float*    s3   = (float*)take((size_t)N * 4);            // layer-3 scalar
  _Float16* Bf   = (_Float16*)take((size_t)8 * 4 * 32 * 16 * 2); // W2 B-fragments
  _Float16* h1h  = (_Float16*)take((size_t)N * HDIM * 2);  // layer-1 output (f16)
  float*    t2   = (float*)take((size_t)N * HDIM * 4);     // h1 @ W2
  float*    acc2 = (float*)take((size_t)N * HDIM * 4);     // layer-2 agg / h2

  const int B = 256;
  const size_t NH = (size_t)N * HDIM;

  // degrees and symmetric norm
  k_fill_deg <<<(N + B - 1) / B, B, 0, stream>>>(dis, N);
  k_accum_deg<<<(E + B - 1) / B, B, 0, stream>>>(dst, dis, E);
  k_make_dis <<<(N + B - 1) / B, B, 0, stream>>>(x, dis, xa, N);

  // layer 1: aggregate scalar x, then expand by W1 (relu), store f16
  k_agg_scalar<<<(E + B - 1) / B, B, 0, stream>>>(src, dst, dis, x, xa, E);
  k_expand_h1 <<<N, HDIM, 0, stream>>>(xa, W1, b1, h1h);

  // layer 2: WMMA GEMM then vector edge aggregation
  k_prep_w2<<<4, 256, 0, stream>>>(W2, Bf);
  k_gemm   <<<N / 16, 256, 0, stream>>>(h1h, Bf, t2, N);
  k_init_acc2<<<(int)((NH + B - 1) / B), B, 0, stream>>>(dis, t2, acc2, NH);
  k_agg_vec<<<(int)(((size_t)E * 32 + B - 1) / B), B, 0, stream>>>(src, dst, dis, t2, acc2, E);
  k_relu_bias<<<(int)((NH + B - 1) / B), B, 0, stream>>>(acc2, b2, NH);

  // layer 3: reduce to scalar per node first, then aggregate scalars
  k_dot_w3 <<<(int)(((size_t)N * 32 + B - 1) / B), B, 0, stream>>>(acc2, W3, s3, N);
  k_init_out<<<(N + B - 1) / B, B, 0, stream>>>(dis, s3, b3, out, N);
  k_agg_scalar<<<(E + B - 1) / B, B, 0, stream>>>(src, dst, dis, s3, out, E);
}
